// GNNStack_42013370089829
// MI455X (gfx1250) — matrix operand order repaired and checked by reading, not compiled, not run on hardware
//
#include <hip/hip_runtime.h>
#include <math.h>

// ---------------------------------------------------------------------------
// GraphSAGE stack for MI455X (gfx1250, wave32).
//  - scatter-mean aggregation: wave-per-edge f32 global atomics (L2-resident)
//  - SAGE linear layers: fused [N,256]x[256,128] GEMM on v_wmma_f32_16x16x32_bf16
//  - layernorm: wave-per-row shuffle reduction
//  - MLP head + log_softmax: plain VALU (negligible FLOPs)
// ---------------------------------------------------------------------------

#define HFEAT 128
#define KTOT  256        // concat(agg, h) along K
#define LDS_STRIDE 132   // dwords per A-tile row (264 bf16) -> bank-conflict-free

typedef __bf16 bf16;
typedef __attribute__((ext_vector_type(16))) __bf16 bf16x16;
typedef __attribute__((ext_vector_type(8)))  float  f32x8;
typedef __attribute__((ext_vector_type(4)))  float  f32x4;

__device__ __forceinline__ unsigned pack_bf16x2(float lo, float hi) {
  // round-to-nearest-even f32 -> bf16, packed (lo in [15:0])
  unsigned ul = __float_as_uint(lo);
  unsigned uh = __float_as_uint(hi);
  unsigned rl = (ul + 0x7FFFu + ((ul >> 16) & 1u)) >> 16;
  unsigned rh = (uh + 0x7FFFu + ((uh >> 16) & 1u)) >> 16;
  return (rl & 0xFFFFu) | (rh << 16);
}

// ---------------------------- degree / scatter -----------------------------

__global__ void k_degree(const int* __restrict__ dst, int E, float* __restrict__ deg) {
  int e = blockIdx.x * blockDim.x + threadIdx.x;
  if (e < E) unsafeAtomicAdd(&deg[dst[e]], 1.0f);
}

__global__ void k_invdeg(const float* __restrict__ deg, float* __restrict__ inv, int N) {
  int i = blockIdx.x * blockDim.x + threadIdx.x;
  if (i < N) inv[i] = 1.0f / fmaxf(deg[i], 1.0f);
}

// one wave per edge; each lane moves 4 contiguous floats (b128 load + 4 atomics)
__global__ __launch_bounds__(256) void k_scatter(const float* __restrict__ x,
                                                 const int* __restrict__ src,
                                                 const int* __restrict__ dst,
                                                 float* __restrict__ agg, int E) {
  int gid  = blockIdx.x * blockDim.x + threadIdx.x;
  int e    = gid >> 5;
  int lane = gid & 31;
  if (e >= E) return;
  int s = src[e], d = dst[e];
  f32x4 v = ((const f32x4*)(x + (size_t)s * HFEAT))[lane];
  float* o = agg + (size_t)d * HFEAT + lane * 4;
  unsafeAtomicAdd(o + 0, v.x);
  unsafeAtomicAdd(o + 1, v.y);
  unsafeAtomicAdd(o + 2, v.z);
  unsafeAtomicAdd(o + 3, v.w);
}

// -------------------------- weight fragment pack ---------------------------
// B-fragment layout (v_wmma_*_bf16, wave32): N = 16*w + lane%16,
// dword d holds K = 8*(lane/16) + 2d (+8 if d>=4), +1.  Index: ((w*8+ks)*32+lane)*8 + d.
// Wcat[n][k] = k<128 ? Wl[n][k] : Wr[n][k-128]   (out = in @ Wcat^T)

__global__ void k_pack_weights(const float* __restrict__ Wl,
                               const float* __restrict__ Wr,
                               unsigned* __restrict__ Bpack) {
  int t = blockIdx.x * blockDim.x + threadIdx.x;    // 0 .. 16383
  if (t >= 8 * 8 * 32 * 8) return;
  int d    = t & 7;
  int lane = (t >> 3) & 31;
  int ks   = (t >> 8) & 7;
  int w    = t >> 11;
  int k  = ks * 32 + ((lane >> 4) << 3) + 2 * d + ((d >> 2) << 3);
  int n0 = w * 16 + (lane & 15);
  const float* W = (k < HFEAT) ? Wl : Wr;
  int kk = k & (HFEAT - 1);
  float lo = W[(size_t)n0 * HFEAT + kk];
  float hi = W[(size_t)n0 * HFEAT + kk + 1];
  Bpack[t] = pack_bf16x2(lo, hi);
}

// ------------------------------- WMMA GEMM ---------------------------------
// out[i,:] = relu( [agg[i]*inv[i] | h[i]] @ Wcat^T + bias )
// Block = 256 threads = 8 waves; one block per 16-node M-tile; each wave owns
// one 16-wide N-tile; K loop = 8 x v_wmma_f32_16x16x32_bf16.

__global__ __launch_bounds__(256) void k_sage_gemm(const float* __restrict__ h,
                                                   const float* __restrict__ agg,
                                                   const float* __restrict__ inv,
                                                   const unsigned* __restrict__ Bpack,
                                                   const float* __restrict__ bias,
                                                   float* __restrict__ out, int nN) {
  __shared__ alignas(16) unsigned ldsA[16 * LDS_STRIDE];  // 16 x 256 bf16 (+pad)

  const int t    = threadIdx.x;
  const int base = blockIdx.x * 16;

  // stage A tile: 2048 bf16 pairs, 8 per thread (blocked -> contiguous loads)
#pragma unroll
  for (int i = 0; i < 8; ++i) {
    int p    = t * 8 + i;
    int row  = p >> 7;          // 0..15
    int pc   = p & 127;         // pair column
    int col0 = pc * 2;          // 0..254
    int node = base + row;
    float a = 0.0f, b = 0.0f;
    if (node < nN) {
      if (col0 < HFEAT) {
        float iv = inv[node];
        const float* ap = agg + (size_t)node * HFEAT + col0;
        a = ap[0] * iv; b = ap[1] * iv;
      } else {
        const float* hp = h + (size_t)node * HFEAT + (col0 - HFEAT);
        a = hp[0]; b = hp[1];
      }
    }
    ldsA[row * LDS_STRIDE + pc] = pack_bf16x2(a, b);
  }
  __syncthreads();

  const int wv   = t >> 5;
  const int lane = t & 31;
  const int m    = lane & 15;
  const int off  = (lane & 16) ? 8 : 0;   // ISA 16-bit A layout: hi lanes hold K+8
  const bf16* abase = (const bf16*)ldsA + m * (LDS_STRIDE * 2);
  const f32x4* bptr = (const f32x4*)(Bpack + (size_t)(wv * 8) * 256) + lane * 2;

  f32x8 acc = {};
#pragma unroll
  for (int ks = 0; ks < 8; ++ks) {
    union { f32x4 f[2]; bf16x16 v; } ua, ub;
    const bf16* ar = abase + ks * 32 + off;
    ua.f[0] = *(const f32x4*)ar;          // K = k0+off   .. +7
    ua.f[1] = *(const f32x4*)(ar + 16);   // K = k0+off+16.. +23
    ub.f[0] = bptr[0];
    ub.f[1] = bptr[1];
    bptr += 64;                           // next 32x16 B fragment (256 dwords)
    acc = __builtin_amdgcn_wmma_f32_16x16x32_bf16(
        /*neg_a=*/false, ua.v, /*neg_b=*/false, ub.v,
        /*c_mod=*/(short)0, acc, /*reuse_a=*/false, /*reuse_b=*/false);
  }

  // C/D layout: lane -> N=wv*16+(lane&15); vgpr r -> M = r + 8*(lane>>4)
  const int n  = wv * 16 + (lane & 15);
  const float bs = bias[n];
  const int rbase = base + ((lane >> 4) << 3);
#pragma unroll
  for (int r = 0; r < 8; ++r) {
    int node = rbase + r;
    if (node < nN) out[(size_t)node * HFEAT + n] = fmaxf(acc[r] + bs, 0.0f);
  }
}

// ------------------------------- layernorm ---------------------------------

__global__ __launch_bounds__(256) void k_layernorm(float* __restrict__ hio,
                                                   const float* __restrict__ w,
                                                   const float* __restrict__ b, int nN) {
  int wave = threadIdx.x >> 5, lane = threadIdx.x & 31;
  int row = blockIdx.x * 8 + wave;
  if (row >= nN) return;
  float* p = hio + (size_t)row * HFEAT;
  float v[4];
  float s = 0.0f;
#pragma unroll
  for (int i = 0; i < 4; ++i) { v[i] = p[lane + i * 32]; s += v[i]; }
#pragma unroll
  for (int msk = 16; msk >= 1; msk >>= 1) s += __shfl_xor(s, msk, 32);
  float mu = s * (1.0f / 128.0f);
  float q = 0.0f;
#pragma unroll
  for (int i = 0; i < 4; ++i) { float d = v[i] - mu; q += d * d; }
#pragma unroll
  for (int msk = 16; msk >= 1; msk >>= 1) q += __shfl_xor(q, msk, 32);
  float rstd = rsqrtf(q * (1.0f / 128.0f) + 1e-5f);
#pragma unroll
  for (int i = 0; i < 4; ++i) {
    int c = lane + i * 32;
    p[c] = (v[i] - mu) * rstd * w[c] + b[c];
  }
}

// ------------------------------- MLP head ----------------------------------

__global__ __launch_bounds__(64) void k_mlp(const float* __restrict__ emb,
                                            const float* __restrict__ W1,
                                            const float* __restrict__ b1,
                                            const float* __restrict__ W2,
                                            const float* __restrict__ b2,
                                            float* __restrict__ outlog, int nN) {
  __shared__ float sh[128];
  __shared__ float sz[50];
  __shared__ float sl[16];
  int node = blockIdx.x;
  if (node >= nN) return;
  int t = threadIdx.x;
  const float* hp = emb + (size_t)node * 128;
  sh[t] = hp[t];
  sh[t + 64] = hp[t + 64];
  __syncthreads();
  if (t < 50) {
    const float* wr = W1 + (size_t)t * 128;
    float acc = b1[t];
#pragma unroll 8
    for (int k = 0; k < 128; ++k) acc += wr[k] * sh[k];
    sz[t] = fmaxf(acc, 0.0f);
  }
  __syncthreads();
  if (t < 16) {
    const float* wr = W2 + (size_t)t * 50;
    float acc = b2[t];
    for (int k = 0; k < 50; ++k) acc += wr[k] * sz[k];
    sl[t] = acc;
  }
  __syncthreads();
  if (t < 16) {
    float mx = sl[0];
    for (int j = 1; j < 16; ++j) mx = fmaxf(mx, sl[j]);
    float se = 0.0f;
    for (int j = 0; j < 16; ++j) se += __expf(sl[j] - mx);
    outlog[(size_t)node * 16 + t] = sl[t] - mx - __logf(se);
  }
}

// ------------------------------- launcher ----------------------------------

extern "C" void kernel_launch(void* const* d_in, const int* in_sizes, int n_in,
                              void* d_out, int out_size, void* d_ws, size_t ws_size,
                              hipStream_t stream) {
  const float* x   = (const float*)d_in[0];
  const int*   ei  = (const int*)d_in[1];
  const float* Wl0 = (const float*)d_in[2];
  const float* bl0 = (const float*)d_in[3];
  const float* Wr0 = (const float*)d_in[4];
  const float* Wl1 = (const float*)d_in[5];
  const float* bl1 = (const float*)d_in[6];
  const float* Wr1 = (const float*)d_in[7];
  const float* Wl2 = (const float*)d_in[8];
  const float* bl2 = (const float*)d_in[9];
  const float* Wr2 = (const float*)d_in[10];
  const float* ln0w = (const float*)d_in[11];
  const float* ln0b = (const float*)d_in[12];
  const float* ln1w = (const float*)d_in[13];
  const float* ln1b = (const float*)d_in[14];
  const float* W1 = (const float*)d_in[15];
  const float* b1 = (const float*)d_in[16];
  const float* W2 = (const float*)d_in[17];
  const float* b2 = (const float*)d_in[18];

  const int N = in_sizes[0] / HFEAT;
  const int E = in_sizes[1] / 2;
  const int* src = ei;
  const int* dst = ei + E;

  // workspace layout
  float*    agg   = (float*)d_ws;                         // N*128
  float*    hbuf  = agg + (size_t)N * HFEAT;              // N*128
  float*    deg   = hbuf + (size_t)N * HFEAT;             // N
  float*    inv   = deg + N;                              // N
  unsigned* Bpack = (unsigned*)(inv + N);                 // 16384 dwords

  float* emb  = (float*)d_out;                 // [N,128]
  float* logp = emb + (size_t)N * HFEAT;       // [N,16]

  const int tiles     = (N + 15) / 16;
  const int scatBlks  = (E * 32 + 255) / 256;
  const size_t aggByt = sizeof(float) * (size_t)N * HFEAT;

  // degrees (dst-only, shared by all layers)
  hipMemsetAsync(deg, 0, sizeof(float) * N, stream);
  k_degree<<<(E + 255) / 256, 256, 0, stream>>>(dst, E, deg);
  k_invdeg<<<(N + 255) / 256, 256, 0, stream>>>(deg, inv, N);

  // ---- layer 0 ----
  hipMemsetAsync(agg, 0, aggByt, stream);
  k_scatter<<<scatBlks, 256, 0, stream>>>(x, src, dst, agg, E);
  k_pack_weights<<<64, 256, 0, stream>>>(Wl0, Wr0, Bpack);
  k_sage_gemm<<<tiles, 256, 0, stream>>>(x, agg, inv, Bpack, bl0, emb, N);
  k_layernorm<<<(N + 7) / 8, 256, 0, stream>>>(emb, ln0w, ln0b, N);

  // ---- layer 1 ----
  hipMemsetAsync(agg, 0, aggByt, stream);
  k_scatter<<<scatBlks, 256, 0, stream>>>(emb, src, dst, agg, E);
  k_pack_weights<<<64, 256, 0, stream>>>(Wl1, Wr1, Bpack);
  k_sage_gemm<<<tiles, 256, 0, stream>>>(emb, agg, inv, Bpack, bl1, hbuf, N);
  k_layernorm<<<(N + 7) / 8, 256, 0, stream>>>(hbuf, ln1w, ln1b, N);

  // ---- layer 2 (no LN) ----
  hipMemsetAsync(agg, 0, aggByt, stream);
  k_scatter<<<scatBlks, 256, 0, stream>>>(hbuf, src, dst, agg, E);
  k_pack_weights<<<64, 256, 0, stream>>>(Wl2, Wr2, Bpack);
  k_sage_gemm<<<tiles, 256, 0, stream>>>(hbuf, agg, inv, Bpack, bl2, emb, N);

  // ---- MLP head + log_softmax ----
  k_mlp<<<N, 64, 0, stream>>>(emb, W1, b1, W2, b2, logp, N);
}
